// AttentionModel_90108413870233
// MI455X (gfx1250) — matrix operand order repaired and checked by reading, not compiled, not run on hardware
//
#include <hip/hip_runtime.h>
#include <hip/hip_bf16.h>
#include <math.h>

#define BB 64
#define AA 8
#define GG 1024
#define DD 32
#define TANH_CLIPV 10.0f
#define NEGV -9000000000000000.0f

typedef __attribute__((ext_vector_type(16))) _Float16 v16h;
typedef __attribute__((ext_vector_type(8)))  _Float16 v8h;
typedef __attribute__((ext_vector_type(8)))  float    v8f;

union V16H { v16h v; v8h h8[2]; };

// A fragment: 16x32 f16 tile, row-major stride 32 halves.
// ISA layout: lane<16 -> row m=lane, halves = K[0..7],K[16..23]; lane>=16 -> K[8..15],K[24..31]
__device__ inline v16h load_a16x32(const _Float16* tile, int lane) {
  int m  = lane & 15;
  int k0 = (lane & 16) ? 8 : 0;
  const _Float16* row = tile + m * 32;
  V16H u;
  u.h8[0] = *(const v8h*)(row + k0);
  u.h8[1] = *(const v8h*)(row + k0 + 16);
  return u.v;
}

// B fragment for Y = X @ W^T : B[k][n] = W[n][k]; Wt points at the 16 rows (neurons) of this n-tile.
// ISA layout: lane<16 -> col n=lane, K[0..7],K[16..23]; lane>=16 -> col n=lane-16, K[8..15],K[24..31]
__device__ inline v16h load_b32x16(const _Float16* Wt, int lane) {
  int n  = lane & 15;
  int k0 = (lane & 16) ? 8 : 0;
  const _Float16* row = Wt + n * 32;
  V16H u;
  u.h8[0] = *(const v8h*)(row + k0);
  u.h8[1] = *(const v8h*)(row + k0 + 16);
  return u.v;
}

__device__ inline v8f wmma_tile(const _Float16* a_lds, const _Float16* Wt, int lane) {
  v16h a = load_a16x32(a_lds, lane);
  v16h b = load_b32x16(Wt, lane);
  v8f c = {};
  return __builtin_amdgcn_wmma_f32_16x16x32_f16(false, a, false, b, (short)0, c, false, false);
}

// D layout: VGPR r, lane l -> row = r + 8*(l>>4), col = (l&15)
__device__ inline void store_tile_lds(float* dst16x32, int ncol0, int lane, v8f c) {
  int n  = (lane & 15) + ncol0;
  int r0 = (lane >> 4) * 8;
#pragma unroll
  for (int r = 0; r < 8; ++r) dst16x32[(r0 + r) * 32 + n] = c[r];
}

// ---------------- init: visited, shift_action, f16 weight conversion ----------------
__global__ __launch_bounds__(256) void k_init(int* visited, float* sa,
    _Float16* wq_h, _Float16* wp1_h, _Float16* wp2_h, _Float16* wnode_h,
    const float* aw_q, const float* aw_p1, const float* aw_p2, const float* W_node) {
  int i = blockIdx.x * blockDim.x + threadIdx.x;
  if (i < BB * GG) visited[i] = 0;
  if (i < BB * AA * AA * 2) sa[i] = ((i % (AA * AA * 2)) == 0) ? 1.0f : 0.0f; // [:,0,0,0]=1
  if (i < 1024) { wq_h[i] = (_Float16)aw_q[i]; wp1_h[i] = (_Float16)aw_p1[i]; wp2_h[i] = (_Float16)aw_p2[i]; }
  if (i < 3072) wnode_h[i] = (_Float16)W_node[i];
}

// ---------------- tiny per-b self-attention producing aeK/aeV ----------------
__global__ __launch_bounds__(64) void k_ae(int agent, const float* sa,
    const float* W_act, const float* b_act,
    const float* sw_q, const float* sb_q, const float* sw_k, const float* sb_k,
    const float* sw_v, const float* sb_v, const float* sw_p1, const float* sb_p1,
    const float* sw_p2, const float* sb_p2, const float* s_ln_g, const float* s_ln_b,
    const float* aw_k, const float* ab_k, const float* aw_v, const float* ab_v,
    float* aeK, float* aeV) {
  int b = blockIdx.x, tid = threadIdx.x;
  __shared__ float x[8 * 32], u[8 * 32], y[8 * 32];
  __shared__ float QKV[3 * 8 * 32];   // Q at +0, K at +256, V at +512
  const float* s0 = sa + b * (AA * AA * 2) + agent * (AA * 2);
  for (int i = tid; i < 256; i += 64) {
    int l = i >> 5, d = i & 31;
    x[i] = s0[l * 2] * W_act[d * 2] + s0[l * 2 + 1] * W_act[d * 2 + 1] + b_act[d];
  }
  __syncthreads();
  for (int p = 0; p < 3; ++p) {
    const float* Wp = (p == 0) ? sw_q : (p == 1) ? sw_k : sw_v;
    const float* Bp = (p == 0) ? sb_q : (p == 1) ? sb_k : sb_v;
    float* outp = QKV + p * 256;
    for (int i = tid; i < 256; i += 64) {
      int l = i >> 5, d = i & 31;
      float s = x[i] + Bp[d];
      const float* wr = Wp + d * 32;
      for (int e = 0; e < 32; ++e) s += x[l * 32 + e] * wr[e];
      u[i] = s;
    }
    __syncthreads();
    if (tid < 8) {
      int l = tid; float m = 0.f, vv = 0.f;
      for (int e = 0; e < 32; ++e) m += u[l * 32 + e];
      m *= (1.f / 32.f);
      for (int e = 0; e < 32; ++e) { float dv = u[l * 32 + e] - m; vv += dv * dv; }
      float inv = rsqrtf(vv * (1.f / 32.f) + 1e-5f);
      for (int e = 0; e < 32; ++e)
        outp[l * 32 + e] = (u[l * 32 + e] - m) * inv * s_ln_g[e] + s_ln_b[e];
    }
    __syncthreads();
  }
  const float* Qt = QKV;
  const float* Kt = QKV + 256;
  const float* Vt = QKV + 512;
  // attention over A=8, scale 1/sqrt(D=32)
  if (tid < 32) {
    int h = tid >> 3, l = tid & 7;
    float sc[8]; float mx = -1e30f;
    for (int m = 0; m < 8; ++m) {
      float s = 0.f;
      for (int e = 0; e < 8; ++e) s += Qt[l * 32 + h * 8 + e] * Kt[m * 32 + h * 8 + e];
      s *= 0.17677669529663687f;
      sc[m] = s; mx = fmaxf(mx, s);
    }
    float se = 0.f;
    for (int m = 0; m < 8; ++m) { sc[m] = __expf(sc[m] - mx); se += sc[m]; }
    float inv = 1.f / se;
    for (int e = 0; e < 8; ++e) {
      float acc = 0.f;
      for (int m = 0; m < 8; ++m) acc += sc[m] * Vt[m * 32 + h * 8 + e];
      y[l * 32 + h * 8 + e] = acc * inv;
    }
  }
  __syncthreads();
  // y2 = relu(y @ sw_p2^T + sb_p2)
  for (int i = tid; i < 256; i += 64) {
    int l = i >> 5, d = i & 31;
    float s = sb_p2[d];
    const float* wr = sw_p2 + d * 32;
    for (int e = 0; e < 32; ++e) s += y[l * 32 + e] * wr[e];
    u[i] = fmaxf(s, 0.f);
  }
  __syncthreads();
  // out = LN(y2 @ sw_p1^T + sb_p1)
  for (int i = tid; i < 256; i += 64) {
    int l = i >> 5, d = i & 31;
    float s = sb_p1[d];
    const float* wr = sw_p1 + d * 32;
    for (int e = 0; e < 32; ++e) s += u[l * 32 + e] * wr[e];
    y[i] = s;
  }
  __syncthreads();
  if (tid < 8) {
    int l = tid; float m = 0.f, vv = 0.f;
    for (int e = 0; e < 32; ++e) m += y[l * 32 + e];
    m *= (1.f / 32.f);
    for (int e = 0; e < 32; ++e) { float dv = y[l * 32 + e] - m; vv += dv * dv; }
    float inv = rsqrtf(vv * (1.f / 32.f) + 1e-5f);
    for (int e = 0; e < 32; ++e)
      x[l * 32 + e] = (y[l * 32 + e] - m) * inv * s_ln_g[e] + s_ln_b[e];
  }
  __syncthreads();
  // aeK = ae @ aw_k^T + ab_k ; aeV likewise
  for (int i = tid; i < 256; i += 64) {
    int l = i >> 5, d = i & 31;
    float sk = ab_k[d], sv = ab_v[d];
    const float* wk = aw_k + d * 32;
    const float* wv = aw_v + d * 32;
    for (int e = 0; e < 32; ++e) { sk += x[l * 32 + e] * wk[e]; sv += x[l * 32 + e] * wv[e]; }
    aeK[b * 256 + i] = sk;
    aeV[b * 256 + i] = sv;
  }
}

// ---------------- fused encoder block: Q-proj, agent-attn, p1, p2, LN, W_node ----------------
__global__ __launch_bounds__(128) void k_enc(int agent, const float* enc_in,
    const _Float16* wq_h, const float* ab_q,
    const _Float16* wp1_h, const float* ab_p1,
    const _Float16* wp2_h, const float* ab_p2,
    const _Float16* wnode_h,
    const float* ln_g, const float* ln_b,
    const float* aeK, const float* aeV,
    float* encO, float* proj, float* gpart) {
  int b = blockIdx.y, g0 = blockIdx.x * 64, tid = threadIdx.x;
  int lane = tid & 31, wave = tid >> 5;
  __shared__ __align__(16) _Float16 Xq[64 * 32];
  __shared__ __align__(16) _Float16 Yh[64 * 32];
  __shared__ __align__(16) _Float16 Y1h[64 * 32];
  __shared__ __align__(16) _Float16 Eh[64 * 32];
  __shared__ float Xr[64 * 32];
  __shared__ float Qf[64 * 32];
  __shared__ float aeKs[256], aeVs[256];
  const float* X = enc_in + ((size_t)(b * AA + agent)) * GG * 32;
  for (int i = tid; i < 2048; i += 128) {
    Xr[i] = X[(size_t)g0 * 32 + i];            // residual rows g0..g0+63
    int c = i >> 9, rem = i & 511, rr = rem >> 5, col = rem & 31;
    int gq = c * 256 + (g0 >> 2) + rr;         // gathered Q-input rows (reshape semantics)
    Xq[i] = (_Float16)X[gq * 32 + col];
  }
  for (int i = tid; i < 256; i += 128) { aeKs[i] = aeK[b * 256 + i]; aeVs[i] = aeV[b * 256 + i]; }
  __syncthreads();
  { // Q = Xq @ aw_q^T   (two 16x16 WMMAs per wave)
    const _Float16* at = Xq + wave * 512;
    v8f c0 = wmma_tile(at, wq_h, lane);
    v8f c1 = wmma_tile(at, wq_h + 512, lane);
    store_tile_lds(Qf + wave * 512, 0, lane, c0);
    store_tile_lds(Qf + wave * 512, 16, lane, c1);
  }
  __syncthreads();
  for (int i = tid; i < 2048; i += 128) Qf[i] += ab_q[i & 31];
  __syncthreads();
  // agent attention over A=8 (scale 1/sqrt(32)); replicates the g-mixing reshape
  for (int task = tid; task < 256; task += 128) {
    int j = task >> 2, nh = task & 3;
    int he = (g0 + j) & 3;
    int qr = nh * 16 + (j >> 2);
    const float* qv = Qf + qr * 32 + he * 8;
    const float* kv = aeKs + he * 8;
    float sc[8]; float mx = -1e30f;
#pragma unroll
    for (int m = 0; m < 8; ++m) {
      float s = 0.f;
#pragma unroll
      for (int e = 0; e < 8; ++e) s += qv[e] * kv[m * 32 + e];
      s *= 0.17677669529663687f;
      sc[m] = s; mx = fmaxf(mx, s);
    }
    float se = 0.f;
#pragma unroll
    for (int m = 0; m < 8; ++m) { sc[m] = __expf(sc[m] - mx); se += sc[m]; }
    float inv = 1.f / se;
    const float* vv = aeVs + he * 8;
#pragma unroll
    for (int e = 0; e < 8; ++e) {
      float acc = 0.f;
#pragma unroll
      for (int m = 0; m < 8; ++m) acc += sc[m] * vv[m * 32 + e];
      Yh[j * 32 + nh * 8 + e] = (_Float16)(acc * inv);
    }
  }
  __syncthreads();
  { // p1
    const _Float16* at = Yh + wave * 512;
    v8f c0 = wmma_tile(at, wp1_h, lane);
    v8f c1 = wmma_tile(at, wp1_h + 512, lane);
    store_tile_lds(Qf + wave * 512, 0, lane, c0);
    store_tile_lds(Qf + wave * 512, 16, lane, c1);
  }
  __syncthreads();
  for (int i = tid; i < 2048; i += 128) Y1h[i] = (_Float16)fmaxf(Qf[i] + ab_p1[i & 31], 0.f);
  __syncthreads();
  { // p2
    const _Float16* at = Y1h + wave * 512;
    v8f c0 = wmma_tile(at, wp2_h, lane);
    v8f c1 = wmma_tile(at, wp2_h + 512, lane);
    store_tile_lds(Qf + wave * 512, 0, lane, c0);
    store_tile_lds(Qf + wave * 512, 16, lane, c1);
  }
  __syncthreads();
  for (int i = tid; i < 2048; i += 128) Xr[i] = Xr[i] + Qf[i] + ab_p2[i & 31];
  __syncthreads();
  // layernorm per row -> enc
  if (tid < 64) {
    int j = tid; float m = 0.f, vv2 = 0.f;
    for (int e = 0; e < 32; ++e) m += Xr[j * 32 + e];
    m *= (1.f / 32.f);
    for (int e = 0; e < 32; ++e) { float dv = Xr[j * 32 + e] - m; vv2 += dv * dv; }
    float inv = rsqrtf(vv2 * (1.f / 32.f) + 1e-5f);
    float* eo = encO + ((size_t)(b * GG + g0 + j)) * 32;
    for (int e = 0; e < 32; ++e) {
      float ev = (Xr[j * 32 + e] - m) * inv * ln_g[e] + ln_b[e];
      Xr[j * 32 + e] = ev;
      Eh[j * 32 + e] = (_Float16)ev;
      eo[e] = ev;
    }
  }
  __syncthreads();
  if (tid < 32) { // deterministic partial graph-sum
    float s = 0.f;
    for (int j = 0; j < 64; ++j) s += Xr[j * 32 + tid];
    gpart[((size_t)b * 16 + blockIdx.x) * 32 + tid] = s;
  }
  { // proj = enc @ W_node^T  (6 n-tiles, no bias), stream to global
    const _Float16* at = Eh + wave * 512;
#pragma unroll
    for (int nt = 0; nt < 6; ++nt) {
      v8f c = wmma_tile(at, wnode_h + nt * 512, lane);
      int n = (lane & 15) + nt * 16;
      int r0 = (lane >> 4) * 8;
#pragma unroll
      for (int r = 0; r < 8; ++r) {
        int g = g0 + wave * 16 + r0 + r;
        proj[((size_t)b * GG + g) * 96 + n] = c[r];
      }
    }
  }
}

// ---------------- query = W_fixed(graph_mean) + W_step(enc[b,idx]) ----------------
__global__ __launch_bounds__(32) void k_query(int agent, const float* gpart, const float* encO,
    const float* W_fixed, const float* W_step, const int* node_last, float* query) {
  int b = blockIdx.x, d = threadIdx.x;
  __shared__ float ge[32], rr[32];
  float s = 0.f;
  for (int c = 0; c < 16; ++c) s += gpart[((size_t)b * 16 + c) * 32 + d];
  ge[d] = s * (1.f / (float)GG);
  int idx = node_last[b * AA + agent];
  rr[d] = encO[((size_t)(b * GG + idx)) * 32 + d];
  __syncthreads();
  float f = 0.f, st = 0.f;
  const float* wf = W_fixed + d * 32;
  const float* wsR = W_step + d * 32;
  for (int e = 0; e < 32; ++e) { f += ge[e] * wf[e]; st += rr[e] * wsR[e]; }
  query[b * 32 + d] = f + st;
}

__device__ inline float blk_max(float v, float* red, int tid) {
  red[tid] = v; __syncthreads();
  for (int s = 128; s > 0; s >>= 1) { if (tid < s) red[tid] = fmaxf(red[tid], red[tid + s]); __syncthreads(); }
  float r = red[0]; __syncthreads();
  return r;
}
__device__ inline float blk_sum(float v, float* red, int tid) {
  red[tid] = v; __syncthreads();
  for (int s = 128; s > 0; s >>= 1) { if (tid < s) red[tid] += red[tid + s]; __syncthreads(); }
  float r = red[0]; __syncthreads();
  return r;
}

// ---------------- final masked attention over G, log-softmax, selection ----------------
__global__ __launch_bounds__(256) void k_select(int agent, const float* proj, const float* query,
    const int* adj, const int* node_last, const float* W_out, const float* node_coords,
    int* visited, float* sa, float* out) {
  int b = blockIdx.x, tid = threadIdx.x;
  __shared__ float wl[4 * GG];
  __shared__ float pacc[256 * 32];
  __shared__ float psum[256 * 4];
  __shared__ float red[256];
  __shared__ int   ridx[256];
  __shared__ float qv[32], score[32], fq[32];
  __shared__ float mh[4], shs[4];
  const float rs8 = 0.35355339059327373f; // 1/sqrt(HD=8)
  int idx = node_last[b * AA + agent];
  const int* arow = adj + (size_t)idx * GG;
  if (tid < 32) qv[tid] = query[b * 32 + tid];
  __syncthreads();
  // w[h][g] with adjacency mask
  for (int g = tid; g < GG; g += 256) {
    const float* gk = proj + ((size_t)b * GG + g) * 96;
    bool ok = arow[g] > 0;
#pragma unroll
    for (int h = 0; h < 4; ++h) {
      float s = NEGV;
      if (ok) {
        s = 0.f;
        for (int e = 0; e < 8; ++e) s += qv[h * 8 + e] * gk[h * 8 + e];
        s *= rs8;
      }
      wl[h * GG + g] = s;
    }
  }
  __syncthreads();
  for (int h = 0; h < 4; ++h) {
    float m = -1e38f;
    for (int g = tid; g < GG; g += 256) m = fmaxf(m, wl[h * GG + g]);
    float M = blk_max(m, red, tid);
    if (tid == 0) mh[h] = M;
    __syncthreads();
  }
  float accl[4][8]; float suml[4];
#pragma unroll
  for (int h = 0; h < 4; ++h) { suml[h] = 0.f;
#pragma unroll
    for (int e = 0; e < 8; ++e) accl[h][e] = 0.f; }
  for (int g = tid; g < GG; g += 256) {
    const float* gv = proj + ((size_t)b * GG + g) * 96 + 32;
#pragma unroll
    for (int h = 0; h < 4; ++h) {
      float ew = __expf(wl[h * GG + g] - mh[h]);
      suml[h] += ew;
      for (int e = 0; e < 8; ++e) accl[h][e] += ew * gv[h * 8 + e];
    }
  }
  for (int t = 0; t < 32; ++t) pacc[tid * 32 + t] = accl[t >> 3][t & 7];
  for (int h = 0; h < 4; ++h) psum[tid * 4 + h] = suml[h];
  __syncthreads();
  if (tid < 32) {
    float s = 0.f;
    for (int t = 0; t < 256; ++t) s += pacc[t * 32 + tid];
    score[tid] = s;
  }
  if (tid < 4) {
    float s = 0.f;
    for (int t = 0; t < 256; ++t) s += psum[t * 4 + tid];
    shs[tid] = s;
  }
  __syncthreads();
  if (tid < 32) score[tid] = score[tid] / shs[tid >> 3];
  __syncthreads();
  if (tid < 32) { // final_Q = score @ W_out^T
    float s = 0.f;
    const float* wr = W_out + tid * 32;
    for (int e = 0; e < 32; ++e) s += score[e] * wr[e];
    fq[tid] = s;
  }
  __syncthreads();
  // logits over G with tanh clip
  for (int g = tid; g < GG; g += 256) {
    const float* lk = proj + ((size_t)b * GG + g) * 96 + 64;
    float s = 0.f;
    for (int e = 0; e < 32; ++e) s += fq[e] * lk[e];
    s *= rs8;
    wl[g] = (arow[g] > 0) ? tanhf(s) * TANH_CLIPV : NEGV;
  }
  __syncthreads();
  float m2 = -1e38f;
  for (int g = tid; g < GG; g += 256) m2 = fmaxf(m2, wl[g]);
  float M2 = blk_max(m2, red, tid);
  float se2 = 0.f;
  for (int g = tid; g < GG; g += 256) se2 += __expf(wl[g] - M2);
  float S2 = blk_sum(se2, red, tid);
  float lse = M2 + logf(S2);
  float slocal = 0.f;
  for (int g = tid; g < GG; g += 256) {
    float lp = wl[g] - lse;
    out[(size_t)b * AA * GG + (size_t)agent * GG + g] = lp;
    float p = __expf(lp);
    slocal += visited[b * GG + g] ? 0.f : p;
  }
  float Spm = blk_sum(slocal, red, tid);
  bool usepm = Spm > 0.f;
  float bestv = -1e38f; int besti = 0;
  for (int g = tid; g < GG; g += 256) {
    float p = __expf(wl[g] - lse);
    float v2 = (usepm && visited[b * GG + g]) ? 0.f : p;
    if (v2 > bestv) { bestv = v2; besti = g; }
  }
  red[tid] = bestv; ridx[tid] = besti; __syncthreads();
  for (int s = 128; s > 0; s >>= 1) {
    if (tid < s) {
      float ov = red[tid + s]; int oi = ridx[tid + s];
      if (ov > red[tid] || (ov == red[tid] && oi < ridx[tid])) { red[tid] = ov; ridx[tid] = oi; }
    }
    __syncthreads();
  }
  int sel = ridx[0];
  if (tid == 0) {
    visited[b * GG + sel] = 1;
    out[(size_t)BB * AA * GG + b * AA + agent] = (float)sel;
  }
  if (agent < AA - 1) {
    float* dst = sa + b * 128 + (agent + 1) * 16;
    const float* src = sa + b * 128 + agent * 16;
    if (tid < 16) dst[tid] = src[tid];
    __syncthreads();
    if (tid == 0) {
      dst[(agent + 1) * 2]     = node_coords[sel * 2];
      dst[(agent + 1) * 2 + 1] = node_coords[sel * 2 + 1];
    }
  }
}

extern "C" void kernel_launch(void* const* d_in, const int* in_sizes, int n_in,
                              void* d_out, int out_size, void* d_ws, size_t ws_size,
                              hipStream_t stream) {
  const float* enc_in      = (const float*)d_in[0];
  const float* node_coords = (const float*)d_in[1];
  const float* aw_k  = (const float*)d_in[2];
  const float* ab_k  = (const float*)d_in[3];
  const float* aw_q  = (const float*)d_in[4];
  const float* ab_q  = (const float*)d_in[5];
  const float* aw_v  = (const float*)d_in[6];
  const float* ab_v  = (const float*)d_in[7];
  const float* aw_p1 = (const float*)d_in[8];
  const float* ab_p1 = (const float*)d_in[9];
  const float* aw_p2 = (const float*)d_in[10];
  const float* ab_p2 = (const float*)d_in[11];
  const float* sw_k  = (const float*)d_in[12];
  const float* sb_k  = (const float*)d_in[13];
  const float* sw_q  = (const float*)d_in[14];
  const float* sb_q  = (const float*)d_in[15];
  const float* sw_v  = (const float*)d_in[16];
  const float* sb_v  = (const float*)d_in[17];
  const float* sw_p1 = (const float*)d_in[18];
  const float* sb_p1 = (const float*)d_in[19];
  const float* sw_p2 = (const float*)d_in[20];
  const float* sb_p2 = (const float*)d_in[21];
  const float* s_ln_g = (const float*)d_in[22];
  const float* s_ln_b = (const float*)d_in[23];
  const float* ln_g   = (const float*)d_in[24];
  const float* ln_b   = (const float*)d_in[25];
  const float* W_node = (const float*)d_in[26];
  const float* W_fixed= (const float*)d_in[27];
  const float* W_step = (const float*)d_in[28];
  const float* W_out  = (const float*)d_in[29];
  const float* W_act  = (const float*)d_in[30];
  const float* b_act  = (const float*)d_in[31];
  const int* adj       = (const int*)d_in[32];
  const int* node_last = (const int*)d_in[33];
  float* out = (float*)d_out;

  char* wsb = (char*)d_ws;
  _Float16* wq_h    = (_Float16*)wsb;
  _Float16* wp1_h   = wq_h + 1024;
  _Float16* wp2_h   = wp1_h + 1024;
  _Float16* wnode_h = wp2_h + 1024;      // 3072 halves
  float* fbase = (float*)(wsb + 12288);
  size_t o = 0;
  float* aeK   = fbase + o; o += (size_t)BB * AA * 32;
  float* aeV   = fbase + o; o += (size_t)BB * AA * 32;
  float* gpart = fbase + o; o += (size_t)BB * 16 * 32;
  float* query = fbase + o; o += (size_t)BB * 32;
  float* saF   = fbase + o; o += (size_t)BB * AA * AA * 2;
  float* encO  = fbase + o; o += (size_t)BB * GG * 32;
  float* proj  = fbase + o; o += (size_t)BB * GG * 96;
  int* visited = (int*)(fbase + o);

  k_init<<<(BB * GG + 255) / 256, 256, 0, stream>>>(visited, saF, wq_h, wp1_h, wp2_h, wnode_h,
                                                    aw_q, aw_p1, aw_p2, W_node);
  for (int agent = 0; agent < AA; ++agent) {
    k_ae<<<BB, 64, 0, stream>>>(agent, saF, W_act, b_act, sw_q, sb_q, sw_k, sb_k, sw_v, sb_v,
                                sw_p1, sb_p1, sw_p2, sb_p2, s_ln_g, s_ln_b,
                                aw_k, ab_k, aw_v, ab_v, aeK, aeV);
    k_enc<<<dim3(GG / 64, BB), 128, 0, stream>>>(agent, enc_in, wq_h, ab_q, wp1_h, ab_p1,
                                                 wp2_h, ab_p2, wnode_h, ln_g, ln_b,
                                                 aeK, aeV, encO, proj, gpart);
    k_query<<<BB, 32, 0, stream>>>(agent, gpart, encO, W_fixed, W_step, node_last, query);
    k_select<<<BB, 256, 0, stream>>>(agent, proj, query, adj, node_last, W_out, node_coords,
                                     visited, saF, out);
  }
  (void)in_sizes; (void)n_in; (void)out_size; (void)ws_size;
}
